// VectorQuantizer_66967130079629
// MI455X (gfx1250) — matrix operand order repaired and checked by reading, not compiled, not run on hardware
//
#include <hip/hip_runtime.h>
#include <hip/hip_bf16.h>

typedef __attribute__((ext_vector_type(16))) _Float16 v16h;
typedef __attribute__((ext_vector_type(8)))  float    v8f;

// Problem dims
#define K_CODES 1024
#define DIM     64
#define N_ROWS  131072          // 32*64*64 flat vectors
#define NUMEL   8388608         // 32*64*64*64

// d_ws layout (float offsets)
#define WS_LOSS     0
#define WS_COUNTS   256         // float[1024]
#define WS_ESUM     1280        // float[65536], 256B-aligned per code row
#define WS_STAGE_F  66816       // staged region begins here (byte 267264, 16B aligned)
#define STAGE_BYTES 135168      // 128KB swizzled f16 B-fragments + 4KB |e|^2
#define LDS_E2_OFF  131072
#define LDS_BYTES   (STAGE_BYTES + 512)   // + 8 waves * 16 ints idx scratch

// d_out layout (float offsets)
#define OUT_Q     0
#define OUT_LOSS  8388608
#define OUT_IDX   8388609
#define OUT_NCS   (8388609 + 131072)
#define OUT_NW    (OUT_NCS + 1024)
#define OUT_NE    (OUT_NW + 65536)

// ---------------------------------------------------------------------------
// Prep: zero accumulators; build swizzled f16 B-fragment table + |e|^2 table.
// B fragment layout (16-bit 32x16, this file's contiguous K order, same as A): lane = h*16+n holds column n,
// K = 16h + e.  Fragment (code tile t, K-chunk c) at byte ((t*2+c)*32+lane)*32.
// ---------------------------------------------------------------------------
__global__ void vq_prep(const float* __restrict__ emb, float* __restrict__ ws) {
  int tid = blockIdx.x * blockDim.x + threadIdx.x;
  if (tid < WS_STAGE_F) ws[tid] = 0.0f;             // loss + counts + emb_sum
  _Float16* st = (_Float16*)(ws + WS_STAGE_F);
  if (tid < 65536) {
    int t = tid >> 10;
    int r = tid & 1023;
    int c = r >> 9;  r &= 511;
    int lane = r >> 4;
    int e = r & 15;
    int h = lane >> 4, n = lane & 15;
    int code = t * 16 + n;
    int d = c * 32 + h * 16 + e;
    st[tid] = (_Float16)emb[code * 64 + d];
  }
  if (tid < K_CODES) {
    const float* er = emb + tid * 64;
    float s = 0.f;
#pragma unroll
    for (int d = 0; d < 64; ++d) s = fmaf(er[d], er[d], s);
    ((float*)((char*)(ws + WS_STAGE_F) + LDS_E2_OFF))[tid] = s;
  }
}

// ---------------------------------------------------------------------------
// Main: per wave, 16 rows vs all 1024 codes with v_wmma_f32_16x16x32_f16.
// Codebook fragments staged once per workgroup into LDS via async LDS loads;
// B fragments double-buffered so ds latency hides under WMMA + argmin VALU.
// ---------------------------------------------------------------------------
__global__ void __launch_bounds__(256)
vq_main(const float* __restrict__ in, const float* __restrict__ emb,
        float* __restrict__ ws, float* __restrict__ out_q,
        float* __restrict__ out_idx) {
  extern __shared__ char smem[];
  const int lane = threadIdx.x & 31;
  const int wave = threadIdx.x >> 5;
  const int h = lane >> 4;       // lane half
  const int n = lane & 15;       // A-row / C-column within half

  // ---- stage 132KB (swizzled f16 codebook + |e|^2) global -> LDS, async ----
  {
    unsigned long long src = (unsigned long long)(ws + WS_STAGE_F);
    for (int i = threadIdx.x; i < STAGE_BYTES / 16; i += 256) {
      unsigned loff = (unsigned)(i * 16);   // LDS byte address (dyn-LDS base 0)
      unsigned goff = loff;                 // matching global byte offset
      asm volatile("global_load_async_to_lds_b128 %0, %1, %2"
                   :: "v"(loff), "v"(goff), "s"(src) : "memory");
    }
  }

  const float* e2lds = (const float*)(smem + LDS_E2_OFF);
  int* idxScratch = (int*)(smem + STAGE_BYTES) + wave * 16;

  for (int it = 0; it < 2; ++it) {
    const int tile = (blockIdx.x * 2 + it) * 8 + wave;   // 16-row tile id
    const int n0 = tile * 16;
    const int b = n0 >> 12;
    const int hw0 = n0 & 4095;                // 16-aligned, never crosses b
    const int base = b * 262144 + hw0;        // float offset of (b, d=0, hw0)

    // ---- A fragments: 16-bit 16x32 layout; lane half h: K = 16h+e
    union { v16h v; _Float16 s[16]; } a0, a1;
    {
      const float* ap = in + base + n;        // this lane's row
#pragma unroll
      for (int e = 0; e < 16; ++e) {
        int d = h * 16 + e;
        a0.s[e] = (_Float16)ap[d * 4096];
        a1.s[e] = (_Float16)ap[(d + 32) * 4096];
      }
    }

    if (it == 0) {
      asm volatile("s_wait_asynccnt 0x0" ::: "memory");
      __syncthreads();                        // staged LDS visible to all waves
    }

    // ---- sweep 64 code tiles; double-buffered B fragments + |e|^2 ----
    float bestS[8]; int bestI[8];
#pragma unroll
    for (int r = 0; r < 8; ++r) { bestS[r] = 3.4e38f; bestI[r] = 0; }

    v16h b0 = *(const v16h*)(smem + 0 * 1024 + lane * 32);
    v16h b1 = *(const v16h*)(smem + 1 * 1024 + lane * 32);
    float e2v = e2lds[n];

#pragma unroll 2
    for (int t = 0; t < 64; ++t) {
      const int tn = (t + 1) & 63;            // prefetch next tile (wraps once)
      const v16h nb0 = *(const v16h*)(smem + (tn * 2 + 0) * 1024 + lane * 32);
      const v16h nb1 = *(const v16h*)(smem + (tn * 2 + 1) * 1024 + lane * 32);
      const float ne2 = e2lds[tn * 16 + n];

      v8f c = {};
      c = __builtin_amdgcn_wmma_f32_16x16x32_f16(false, a0.v, false, b0,
                                                 (short)0, c, false, false);
      c = __builtin_amdgcn_wmma_f32_16x16x32_f16(false, a1.v, false, b1,
                                                 (short)0, c, false, false);
      const int code = t * 16 + n;
#pragma unroll
      for (int r = 0; r < 8; ++r) {
        float s = fmaf(-2.0f, c[r], e2v);     // ||x||^2 constant per row: drop
        if (s < bestS[r]) { bestS[r] = s; bestI[r] = code; }
      }
      b0 = nb0; b1 = nb1; e2v = ne2;
    }

    // ---- argmin across the 16 lanes of each half (C column = lane%16) ----
#pragma unroll
    for (int m = 1; m < 16; m <<= 1) {
#pragma unroll
      for (int r = 0; r < 8; ++r) {
        float os = __shfl_xor(bestS[r], m, 32);
        int   oi = __shfl_xor(bestI[r], m, 32);
        if (os < bestS[r] || (os == bestS[r] && oi < bestI[r])) {
          bestS[r] = os; bestI[r] = oi;
        }
      }
    }

    // rows 8h+r now resolved in every lane of half h; park 16 idx in LDS
    if (n == 0) {
#pragma unroll
      for (int r = 0; r < 8; ++r) idxScratch[h * 8 + r] = bestI[r];
    }
    asm volatile("s_wait_dscnt 0x0" ::: "memory");  // wave-local store->load

    // ---- quantized output + loss (coalesced along rows) ----
    const int row = n;                        // lanes h=0/1 split even/odd d
    const int myIdx = idxScratch[row];
    const float* qrow = emb + myIdx * 64;
    float lossAcc = 0.f;
#pragma unroll 4
    for (int i = 0; i < 32; ++i) {
      const int d = i * 2 + h;
      const int off = base + d * 4096 + row;
      float x = in[off];
      float q = qrow[d];
      out_q[off] = q;                         // straight-through == quantized
      float diff = q - x;
      lossAcc = fmaf(diff, diff, lossAcc);
    }
    if (lane < 16) {
      atomicAdd(ws + WS_COUNTS + myIdx, 1.0f);
      out_idx[n0 + row] = (float)myIdx;
    }
#pragma unroll
    for (int m = 16; m >= 1; m >>= 1) lossAcc += __shfl_xor(lossAcc, m, 32);
    if (lane == 0) atomicAdd(ws + WS_LOSS, lossAcc);

    // ---- EMA scatter, row-serial so each atomic instr hits ONE code row ----
    // lanes span d: 32 lanes -> 128 contiguous bytes per instruction (code
    // rows are 256B-aligned), ~16x fewer L2 atomic line transactions than a
    // row-per-lane layout.  The transposed x reloads re-hit the same 64B
    // segments 16x, so they are L0/L2 hits.
#pragma unroll 4
    for (int i = 0; i < 16; ++i) {
      const int ci = idxScratch[i];           // code for row i (wave-uniform)
      float x0 = in[base + lane * 4096 + i];         // d = lane
      float x1 = in[base + (lane + 32) * 4096 + i];  // d = lane + 32
      atomicAdd(ws + WS_ESUM + ci * 64 + lane, x0);
      atomicAdd(ws + WS_ESUM + ci * 64 + lane + 32, x1);
    }
  }
}

// ---------------------------------------------------------------------------
// Finalize: EMA updates + normalized embeddings + vq_loss (1 block, K threads)
// ---------------------------------------------------------------------------
__global__ void __launch_bounds__(1024)
vq_finalize(const float* __restrict__ ema_cs, const float* __restrict__ ema_w,
            const float* __restrict__ ws, float* __restrict__ out) {
  __shared__ float red[1024];
  const int k = threadIdx.x;
  float ncs = ema_cs[k] * 0.99f + 0.01f * ws[WS_COUNTS + k];
  out[OUT_NCS + k] = ncs;
  red[k] = ncs;
  __syncthreads();
  for (int s = 512; s > 0; s >>= 1) {
    if (k < s) red[k] += red[k + s];
    __syncthreads();
  }
  float nsum = red[0];
  float cs = (ncs + 1e-5f) / (nsum + 1024.f * 1e-5f) * nsum;
  float inv = 1.0f / cs;
#pragma unroll 8
  for (int d = 0; d < 64; ++d) {
    float w = ema_w[k * 64 + d] * 0.99f + 0.01f * ws[WS_ESUM + k * 64 + d];
    out[OUT_NW + k * 64 + d] = w;
    out[OUT_NE + k * 64 + d] = w * inv;
  }
  if (k == 0) out[OUT_LOSS] = 1.25f * ws[WS_LOSS] / 8388608.0f;
}

extern "C" void kernel_launch(void* const* d_in, const int* in_sizes, int n_in,
                              void* d_out, int out_size, void* d_ws, size_t ws_size,
                              hipStream_t stream) {
  (void)in_sizes; (void)n_in; (void)out_size; (void)ws_size;
  const float* in   = (const float*)d_in[0];   // [32,64,64,64]
  const float* emb  = (const float*)d_in[1];   // [1024,64]
  const float* ecs  = (const float*)d_in[2];   // [1024]
  const float* emaw = (const float*)d_in[3];   // [1024,64]
  float* out = (float*)d_out;
  float* ws  = (float*)d_ws;

  vq_prep<<<261, 256, 0, stream>>>(emb, ws);                    // 66816 threads
  vq_main<<<512, 256, LDS_BYTES, stream>>>(in, emb, ws,
                                           out + OUT_Q, out + OUT_IDX);
  vq_finalize<<<1, 1024, 0, stream>>>(ecs, emaw, ws, out);
}
